// DiffrateEncoderLayer_38706245272212
// MI455X (gfx1250) — matrix operand order repaired and checked by reading, not compiled, not run on hardware
//
#include <hip/hip_runtime.h>
#include <hip/hip_bf16.h>
#include <math.h>

// ---------------- problem constants ----------------
constexpr int Bd   = 128;
constexpr int Nd   = 197;
constexpr int Dd   = 768;
constexpr int Hh   = 12;
constexpr int HD   = 64;
constexpr int DFF  = 3072;
constexpr int PRUNE = 160;
constexpr int MERGE = 120;
constexpr int UNI   = 40;          // PRUNE - MERGE
constexpr int NPAD  = 208;         // token pad (mult of 16)
constexpr int KPAD  = 224;         // key pad for P·v (mult of 32)
constexpr int MPAD2 = 128;         // merged-token pad (mult of 16)
constexpr float LN_EPS = 1e-5f;

typedef __attribute__((ext_vector_type(16))) __bf16 v16bf;
typedef __attribute__((ext_vector_type(8)))  float  v8f;

// ---------------- helpers ----------------
__device__ __forceinline__ unsigned short f2bf(float f) {
  unsigned int u = __float_as_uint(f);
  unsigned int r = u + 0x7FFFu + ((u >> 16) & 1u);   // round-to-nearest-even
  return (unsigned short)(r >> 16);
}

// ---- single 16x16 tile (used by the small attention GEMMs) ----
__device__ __forceinline__ v8f wmma_tile(const unsigned short* A, int lda,
                                         const unsigned short* Bm, int ldb,
                                         int K, int lane) {
  const int r  = lane & 15;
  const int hi = lane >> 4;
  const unsigned short* ap = A  + (size_t)r * lda + hi * 8;
  const unsigned short* bp = Bm + (size_t)r * ldb + hi * 8;
  v8f c = {0.f,0.f,0.f,0.f,0.f,0.f,0.f,0.f};
  for (int k = 0; k < K; k += 32) {
    union { uint4 q[2]; v16bf v; } a, b;
    a.q[0] = *(const uint4*)(ap + k);
    a.q[1] = *(const uint4*)(ap + k + 16);
    b.q[0] = *(const uint4*)(bp + k);
    b.q[1] = *(const uint4*)(bp + k + 16);
    c = __builtin_amdgcn_wmma_f32_16x16x32_bf16(false, a.v, false, b.v,
                                                (short)0, c, false, false);
  }
  return c;
}

// ---- 2x4 register-blocked 32x64 macro-tile ----
// 8 accumulators (64 VGPRs) + 6 fragments (48 VGPRs) per k-step: fits well
// under the 256 directly-addressable VGPRs, so no scratch spills; 8 WMMAs per
// 12 b128 loads (A-frags reused 4x, B-frags 2x).
__device__ __forceinline__ void wmma_tile24(const unsigned short* A, int lda,
                                            const unsigned short* Bm, int ldb,
                                            int K, int lane, v8f c[2][4]) {
  const int r  = lane & 15;
  const int hi = lane >> 4;
  const unsigned short* ap = A  + (size_t)r * lda + hi * 8;
  const unsigned short* bp = Bm + (size_t)r * ldb + hi * 8;
  for (int k = 0; k < K; k += 32) {
    union U { uint4 q[2]; v16bf v; } a[2], b[4];
    #pragma unroll
    for (int i = 0; i < 2; ++i) {
      const unsigned short* pa = ap + (size_t)i * 16 * lda + k;
      a[i].q[0] = *(const uint4*)(pa);
      a[i].q[1] = *(const uint4*)(pa + 16);
    }
    #pragma unroll
    for (int i = 0; i < 4; ++i) {
      const unsigned short* pb = bp + (size_t)i * 16 * ldb + k;
      b[i].q[0] = *(const uint4*)(pb);
      b[i].q[1] = *(const uint4*)(pb + 16);
    }
    #pragma unroll
    for (int mi = 0; mi < 2; ++mi)
      #pragma unroll
      for (int ni = 0; ni < 4; ++ni)
        c[mi][ni] = __builtin_amdgcn_wmma_f32_16x16x32_bf16(
            false, a[mi].v, false, b[ni].v, (short)0, c[mi][ni], false, false);
  }
}

__device__ __forceinline__ void zero24(v8f c[2][4]) {
  v8f z = {0.f,0.f,0.f,0.f,0.f,0.f,0.f,0.f};
  #pragma unroll
  for (int mi = 0; mi < 2; ++mi)
    #pragma unroll
    for (int ni = 0; ni < 4; ++ni) c[mi][ni] = z;
}

// ---------------- kernels ----------------
__global__ void cast_bf16(const float* __restrict__ src, unsigned short* dst, int n) {
  int i = blockIdx.x * 256 + threadIdx.x;
  if (i < n) dst[i] = f2bf(src[i]);
}

__global__ void ln1_kernel(const float* __restrict__ hs, const float* __restrict__ w,
                           const float* __restrict__ bia, unsigned short* xbf) {
  int wave = threadIdx.x >> 5, lane = threadIdx.x & 31;
  int row = blockIdx.x * 8 + wave;
  if (row >= Bd * NPAD) return;
  int b = row / NPAD, i = row % NPAD;
  unsigned short* dst = xbf + (size_t)row * Dd;
  if (i >= Nd) { for (int e = lane; e < Dd; e += 32) dst[e] = 0; return; }
  const float* src = hs + ((size_t)b * Nd + i) * Dd;
  float vals[24]; float s = 0.f;
  for (int t = 0; t < 24; ++t) { vals[t] = src[lane + t * 32]; s += vals[t]; }
  for (int m = 16; m; m >>= 1) s += __shfl_xor(s, m, 32);
  float mu = s * (1.f / Dd);
  float vs = 0.f;
  for (int t = 0; t < 24; ++t) { float d = vals[t] - mu; vs += d * d; }
  for (int m = 16; m; m >>= 1) vs += __shfl_xor(vs, m, 32);
  float rstd = rsqrtf(vs * (1.f / Dd) + LN_EPS);
  for (int t = 0; t < 24; ++t) {
    int e = lane + t * 32;
    dst[e] = f2bf((vals[t] - mu) * rstd * w[e] + bia[e]);
  }
}

__global__ void gemm_qkv(const unsigned short* __restrict__ x,
                         const unsigned short* __restrict__ wqb,
                         const unsigned short* __restrict__ wkb,
                         const unsigned short* __restrict__ wvb,
                         const float* __restrict__ bq, const float* __restrict__ bk,
                         const float* __restrict__ bv,
                         unsigned short* qo, unsigned short* ko, unsigned short* vT) {
  int lane = threadIdx.x & 31;
  long tile = (long)blockIdx.x * 8 + (threadIdx.x >> 5);
  const int NT = Dd / 64;                              // 12
  const long total = (long)(Bd * NPAD / 32) * NT;      // 832*12
  if (tile >= total) return;
  int tm = (int)(tile / NT), tn = (int)(tile % NT);
  int mode = blockIdx.y;
  const unsigned short* wsel = (mode == 0) ? wqb : (mode == 1) ? wkb : wvb;
  v8f c[2][4]; zero24(c);
  wmma_tile24(x + (size_t)tm * 32 * Dd, Dd, wsel + (size_t)tn * 64 * Dd, Dd, Dd, lane, c);
  int n = lane & 15, hi = lane >> 4;
  #pragma unroll
  for (int mi = 0; mi < 2; ++mi)
    #pragma unroll
    for (int ni = 0; ni < 4; ++ni) {
      int o = tn * 64 + ni * 16 + n;
      #pragma unroll
      for (int vv = 0; vv < 8; ++vv) {
        int t = tm * 32 + mi * 16 + vv + hi * 8;
        int b = t / NPAD, i = t % NPAD;
        float val = c[mi][ni][vv];
        if (mode == 0) {
          qo[(size_t)t * Dd + o] = f2bf((val + bq[o]) * 0.125f);   // *HD^-0.5
        } else if (mode == 1) {
          ko[(size_t)t * Dd + o] = f2bf(val + bk[o]);
        } else {
          int hh = o >> 6, hd = o & 63;
          vT[(((size_t)b * Hh + hh) * HD + hd) * KPAD + i] = f2bf(val + bv[o]);
        }
      }
    }
}

__global__ void gemm_scores(const unsigned short* __restrict__ q,
                            const unsigned short* __restrict__ k, float* attn) {
  int lane = threadIdx.x & 31;
  long tile = (long)blockIdx.x * 8 + (threadIdx.x >> 5);
  const int TPG = 13 * 13;
  const long total = (long)Bd * Hh * TPG;
  if (tile >= total) return;
  int bh = (int)(tile / TPG), r = (int)(tile % TPG);
  int tm = r / 13, tn = r % 13;
  int b = bh / Hh, hh = bh % Hh;
  const unsigned short* A  = q + ((size_t)b * NPAD + tm * 16) * Dd + hh * HD;
  const unsigned short* Bm = k + ((size_t)b * NPAD + tn * 16) * Dd + hh * HD;
  v8f c = wmma_tile(A, Dd, Bm, Dd, HD, lane);
  int n = lane & 15, hi = lane >> 4;
  int j = tn * 16 + n;
  if (j < Nd) {
    for (int vv = 0; vv < 8; ++vv) {
      int i = tm * 16 + vv + hi * 8;
      if (i < Nd) attn[((size_t)bh * Nd + i) * Nd + j] = c[vv];
    }
  }
}

__global__ void softmax_kernel(float* attn, unsigned short* P) {
  int wave = threadIdx.x >> 5, lane = threadIdx.x & 31;
  long row = (long)blockIdx.x * 8 + wave;
  const long rows = (long)Bd * Hh * Nd;
  if (row >= rows) return;
  float* s = attn + row * Nd;
  long bh = row / Nd; int i = (int)(row % Nd);
  unsigned short* p = P + ((size_t)bh * NPAD + i) * KPAD;
  float v[7]; float mx = -1e30f;
  for (int t = 0; t < 7; ++t) { int e = lane + t * 32; v[t] = (e < Nd) ? s[e] : -1e30f; mx = fmaxf(mx, v[t]); }
  for (int m = 16; m; m >>= 1) mx = fmaxf(mx, __shfl_xor(mx, m, 32));
  float sum = 0.f;
  for (int t = 0; t < 7; ++t) { int e = lane + t * 32; if (e < Nd) { v[t] = __expf(v[t] - mx); sum += v[t]; } }
  for (int m = 16; m; m >>= 1) sum += __shfl_xor(sum, m, 32);
  float rs = 1.f / sum;
  for (int t = 0; t < 7; ++t) {
    int e = lane + t * 32;
    if (e < Nd) { float pr = v[t] * rs; s[e] = pr; p[e] = f2bf(pr); }
    else if (e < KPAD) p[e] = 0;     // zero-pad keys 197..223
  }
}

__global__ void gemm_av(const unsigned short* __restrict__ P,
                        const unsigned short* __restrict__ vT, unsigned short* ao) {
  int lane = threadIdx.x & 31;
  long tile = (long)blockIdx.x * 8 + (threadIdx.x >> 5);
  const int TPG = 13 * 4;
  const long total = (long)Bd * Hh * TPG;
  if (tile >= total) return;
  int bh = (int)(tile / TPG), r = (int)(tile % TPG);
  int tm = r / 4, tn = r % 4;
  const unsigned short* A  = P  + ((size_t)bh * NPAD + tm * 16) * KPAD;
  const unsigned short* Bm = vT + ((size_t)bh * HD   + tn * 16) * KPAD;
  v8f c = wmma_tile(A, KPAD, Bm, KPAD, KPAD, lane);
  int b = bh / Hh, hh = bh % Hh;
  int n = lane & 15, hi = lane >> 4;
  for (int vv = 0; vv < 8; ++vv) {
    int i = tm * 16 + vv + hi * 8;
    ao[((size_t)b * NPAD + i) * Dd + hh * HD + tn * 16 + n] = f2bf(c[vv]);
  }
}

__global__ void gemm_o(const unsigned short* __restrict__ ao,
                       const unsigned short* __restrict__ wob,
                       const float* __restrict__ bo, const float* __restrict__ hs,
                       float* hfull) {
  int lane = threadIdx.x & 31;
  long tile = (long)blockIdx.x * 8 + (threadIdx.x >> 5);
  const int NT = Dd / 64;
  const long total = (long)(Bd * NPAD / 32) * NT;
  if (tile >= total) return;
  int tm = (int)(tile / NT), tn = (int)(tile % NT);
  v8f c[2][4]; zero24(c);
  wmma_tile24(ao + (size_t)tm * 32 * Dd, Dd, wob + (size_t)tn * 64 * Dd, Dd, Dd, lane, c);
  int n = lane & 15, hi = lane >> 4;
  #pragma unroll
  for (int mi = 0; mi < 2; ++mi)
    #pragma unroll
    for (int ni = 0; ni < 4; ++ni) {
      int o = tn * 64 + ni * 16 + n;
      #pragma unroll
      for (int vv = 0; vv < 8; ++vv) {
        int t = tm * 32 + mi * 16 + vv + hi * 8;
        int b = t / NPAD, i = t % NPAD;
        if (i < Nd) {
          size_t idx = ((size_t)b * Nd + i) * Dd + o;
          hfull[idx] = c[mi][ni][vv] + bo[o] + hs[idx];      // residual
        }
      }
    }
}

__global__ void prune_gather(const float* __restrict__ attn,
                             const float* __restrict__ hfull, float* hp) {
  __shared__ float cls[196];
  __shared__ int perm[PRUNE];
  int b = blockIdx.x, tid = threadIdx.x;
  for (int j = tid; j < 196; j += 256) {
    float s = 0.f;
    for (int hh = 0; hh < Hh; ++hh) s += attn[((size_t)(b * Hh + hh) * Nd) * Nd + (j + 1)];
    cls[j] = s * (1.f / Hh);
  }
  __syncthreads();
  if (tid == 0) perm[0] = 0;
  for (int j = tid; j < 196; j += 256) {
    float cj = cls[j];
    int rank = 0;
    for (int t = 0; t < 196; ++t) {
      float ct = cls[t];
      rank += (ct > cj) || (ct == cj && t < j);   // stable argsort(-cls)
    }
    if (rank < PRUNE - 1) perm[rank + 1] = j + 1;
  }
  __syncthreads();
  for (int e = tid; e < PRUNE * Dd; e += 256) {
    int r = e / Dd, d = e % Dd;
    hp[((size_t)b * PRUNE + r) * Dd + d] = hfull[((size_t)b * Nd + perm[r]) * Dd + d];
  }
}

__global__ void rownorm(const float* __restrict__ hp, float* inv) {
  int wave = threadIdx.x >> 5, lane = threadIdx.x & 31;
  int row = blockIdx.x * 8 + wave;
  if (row >= Bd * PRUNE) return;
  const float* src = hp + (size_t)row * Dd;
  float s = 0.f;
  for (int t = 0; t < 24; ++t) { float x = src[lane + t * 32]; s += x * x; }
  for (int m = 16; m; m >>= 1) s += __shfl_xor(s, m, 32);
  if (lane == 0) inv[row] = rsqrtf(s);
}

__global__ void sim_argmax(const float* __restrict__ hp, const float* __restrict__ inv,
                           int* ms) {
  __shared__ float urow[Dd];
  __shared__ float bvs[8]; __shared__ int bis[8];
  int b = blockIdx.x / UNI, u = blockIdx.x % UNI;
  int tid = threadIdx.x, wave = tid >> 5, lane = tid & 31;
  const float* up = hp + ((size_t)b * PRUNE + MERGE + u) * Dd;
  for (int e = tid; e < Dd; e += 256) urow[e] = up[e];
  __syncthreads();
  float invu = inv[b * PRUNE + MERGE + u];
  float best = -1e30f; int bidx = 0;
  for (int m = wave; m < MERGE; m += 8) {
    const float* mp = hp + ((size_t)b * PRUNE + m) * Dd;
    float s = 0.f;
    for (int t = 0; t < 24; ++t) { int e = lane + t * 32; s += mp[e] * urow[e]; }
    for (int mm = 16; mm; mm >>= 1) s += __shfl_xor(s, mm, 32);
    float val = (m == 0) ? -1e30f : s * invu * inv[b * PRUNE + m];  // sim[...,0]=-inf
    if (val > best || (val == best && m < bidx)) { best = val; bidx = m; }
  }
  if (lane == 0) { bvs[wave] = best; bis[wave] = bidx; }
  __syncthreads();
  if (tid == 0) {
    float bb = bvs[0]; int ii = bis[0];
    for (int w = 1; w < 8; ++w)
      if (bvs[w] > bb || (bvs[w] == bb && bis[w] < ii)) { bb = bvs[w]; ii = bis[w]; }
    ms[b * UNI + u] = ii;
  }
}

__global__ void merge_kernel(const float* __restrict__ hp, const int* __restrict__ ms,
                             float* merged) {
  __shared__ int msl[UNI];
  __shared__ float cnt[MERGE];
  int b = blockIdx.x, tid = threadIdx.x;
  if (tid < MERGE) cnt[tid] = 1.f;
  __syncthreads();
  if (tid < UNI) { int m = ms[b * UNI + tid]; msl[tid] = m; atomicAdd(&cnt[m], 1.f); }
  __syncthreads();
  for (int e = tid; e < MERGE * Dd; e += 256) {
    int m = e / Dd, d = e % Dd;
    float acc = hp[((size_t)b * PRUNE + m) * Dd + d];
    for (int u = 0; u < UNI; ++u)
      if (msl[u] == m) acc += hp[((size_t)b * PRUNE + MERGE + u) * Dd + d];
    merged[((size_t)b * MERGE + m) * Dd + d] = acc / cnt[m];
  }
}

__global__ void ln2_kernel(const float* __restrict__ merged, const float* __restrict__ w,
                           const float* __restrict__ bia, unsigned short* ybf) {
  int wave = threadIdx.x >> 5, lane = threadIdx.x & 31;
  int row = blockIdx.x * 8 + wave;
  if (row >= Bd * MPAD2) return;
  int b = row / MPAD2, i = row % MPAD2;
  unsigned short* dst = ybf + (size_t)row * Dd;
  if (i >= MERGE) { for (int e = lane; e < Dd; e += 32) dst[e] = 0; return; }
  const float* src = merged + ((size_t)b * MERGE + i) * Dd;
  float vals[24]; float s = 0.f;
  for (int t = 0; t < 24; ++t) { vals[t] = src[lane + t * 32]; s += vals[t]; }
  for (int m = 16; m; m >>= 1) s += __shfl_xor(s, m, 32);
  float mu = s * (1.f / Dd);
  float vs = 0.f;
  for (int t = 0; t < 24; ++t) { float d = vals[t] - mu; vs += d * d; }
  for (int m = 16; m; m >>= 1) vs += __shfl_xor(vs, m, 32);
  float rstd = rsqrtf(vs * (1.f / Dd) + LN_EPS);
  for (int t = 0; t < 24; ++t) {
    int e = lane + t * 32;
    dst[e] = f2bf((vals[t] - mu) * rstd * w[e] + bia[e]);
  }
}

__global__ void gemm_ffn1(const unsigned short* __restrict__ y,
                          const unsigned short* __restrict__ w1b,
                          const float* __restrict__ b1, unsigned short* ff) {
  int lane = threadIdx.x & 31;
  long tile = (long)blockIdx.x * 8 + (threadIdx.x >> 5);
  const int NT = DFF / 64;                         // 48
  const long total = (long)(Bd * MPAD2 / 32) * NT; // 512*48
  if (tile >= total) return;
  int tm = (int)(tile / NT), tn = (int)(tile % NT);
  v8f c[2][4]; zero24(c);
  wmma_tile24(y + (size_t)tm * 32 * Dd, Dd, w1b + (size_t)tn * 64 * Dd, Dd, Dd, lane, c);
  int n = lane & 15, hi = lane >> 4;
  #pragma unroll
  for (int mi = 0; mi < 2; ++mi)
    #pragma unroll
    for (int ni = 0; ni < 4; ++ni) {
      int o = tn * 64 + ni * 16 + n;
      #pragma unroll
      for (int vv = 0; vv < 8; ++vv) {
        int t = tm * 32 + mi * 16 + vv + hi * 8;
        float g = c[mi][ni][vv] + b1[o];
        float act = g / (1.f + __expf(-1.702f * g));    // quick-GELU
        ff[(size_t)t * DFF + o] = f2bf(act);
      }
    }
}

__global__ void gemm_ffn2(const unsigned short* __restrict__ ff,
                          const unsigned short* __restrict__ w2b,
                          const float* __restrict__ b2, const float* __restrict__ merged,
                          float* outy) {
  int lane = threadIdx.x & 31;
  long tile = (long)blockIdx.x * 8 + (threadIdx.x >> 5);
  const int NT = Dd / 64;                          // 12
  const long total = (long)(Bd * MPAD2 / 32) * NT; // 512*12
  if (tile >= total) return;
  int tm = (int)(tile / NT), tn = (int)(tile % NT);
  v8f c[2][4]; zero24(c);
  wmma_tile24(ff + (size_t)tm * 32 * DFF, DFF, w2b + (size_t)tn * 64 * DFF, DFF, DFF, lane, c);
  int n = lane & 15, hi = lane >> 4;
  #pragma unroll
  for (int mi = 0; mi < 2; ++mi)
    #pragma unroll
    for (int ni = 0; ni < 4; ++ni) {
      int o = tn * 64 + ni * 16 + n;
      #pragma unroll
      for (int vv = 0; vv < 8; ++vv) {
        int t = tm * 32 + mi * 16 + vv + hi * 8;
        int b = t / MPAD2, i = t % MPAD2;
        if (i < MERGE) {
          size_t idx = ((size_t)b * MERGE + i) * Dd + o;
          outy[idx] = c[mi][ni][vv] + b2[o] + merged[idx];       // residual
        }
      }
    }
}

// ---------------- launcher ----------------
extern "C" void kernel_launch(void* const* d_in, const int* in_sizes, int n_in,
                              void* d_out, int out_size, void* d_ws, size_t ws_size,
                              hipStream_t stream) {
  (void)in_sizes; (void)n_in; (void)out_size; (void)ws_size;
  const float* hs   = (const float*)d_in[0];
  const float* ln1w = (const float*)d_in[1];
  const float* ln1b = (const float*)d_in[2];
  const float* wq   = (const float*)d_in[3];
  const float* bq   = (const float*)d_in[4];
  const float* wk   = (const float*)d_in[5];
  const float* bk   = (const float*)d_in[6];
  const float* wv   = (const float*)d_in[7];
  const float* bv   = (const float*)d_in[8];
  const float* wo   = (const float*)d_in[9];
  const float* bo   = (const float*)d_in[10];
  const float* ln2w = (const float*)d_in[11];
  const float* ln2b = (const float*)d_in[12];
  const float* w1   = (const float*)d_in[13];
  const float* b1   = (const float*)d_in[14];
  const float* w2   = (const float*)d_in[15];
  const float* b2   = (const float*)d_in[16];

  float* out_y = (float*)d_out;                                  // (B,120,768)
  float* attn  = (float*)d_out + (size_t)Bd * MERGE * Dd;        // (B,12,197,197)

  char* ws = (char*)d_ws;
  size_t off = 0;
  auto take = [&](size_t bytes) -> char* {
    char* p = ws + off; off += (bytes + 255) & ~(size_t)255; return p;
  };
  unsigned short* X_BF = (unsigned short*)take((size_t)Bd * NPAD * Dd * 2);
  unsigned short* WQ   = (unsigned short*)take((size_t)Dd * Dd * 2);
  unsigned short* WK   = (unsigned short*)take((size_t)Dd * Dd * 2);
  unsigned short* WV   = (unsigned short*)take((size_t)Dd * Dd * 2);
  unsigned short* WO   = (unsigned short*)take((size_t)Dd * Dd * 2);
  unsigned short* W1   = (unsigned short*)take((size_t)DFF * Dd * 2);
  unsigned short* W2   = (unsigned short*)take((size_t)Dd * DFF * 2);
  unsigned short* Q_BF = (unsigned short*)take((size_t)Bd * NPAD * Dd * 2);
  unsigned short* K_BF = (unsigned short*)take((size_t)Bd * NPAD * Dd * 2);
  unsigned short* VT   = (unsigned short*)take((size_t)Bd * Hh * HD * KPAD * 2);
  unsigned short* P_BF = (unsigned short*)take((size_t)Bd * Hh * NPAD * KPAD * 2);
  float* HFULL  = (float*)take((size_t)Bd * Nd * Dd * 4);
  float* MERGED = (float*)take((size_t)Bd * MERGE * Dd * 4);
  float* INV    = (float*)take((size_t)Bd * PRUNE * 4);
  int*   MS     = (int*)take((size_t)Bd * UNI * 4);
  // lifetime-safe aliases
  unsigned short* AO  = X_BF;                  // attn_out reuses x (x dead after QKV)
  unsigned short* YBF = Q_BF;                  // ln2 output reuses q (dead after scores)
  float*          HP  = (float*)K_BF;          // pruned h reuses k+vT (62.9MB <= 84.9MB)
  unsigned short* FF  = P_BF;                  // ffn hidden reuses P (100.7MB <= 143.1MB)

  const dim3 blk(256);

  // weights -> bf16
  cast_bf16<<<dim3((Dd*Dd + 255)/256), blk, 0, stream>>>(wq, WQ, Dd*Dd);
  cast_bf16<<<dim3((Dd*Dd + 255)/256), blk, 0, stream>>>(wk, WK, Dd*Dd);
  cast_bf16<<<dim3((Dd*Dd + 255)/256), blk, 0, stream>>>(wv, WV, Dd*Dd);
  cast_bf16<<<dim3((Dd*Dd + 255)/256), blk, 0, stream>>>(wo, WO, Dd*Dd);
  cast_bf16<<<dim3((DFF*Dd + 255)/256), blk, 0, stream>>>(w1, W1, DFF*Dd);
  cast_bf16<<<dim3((Dd*DFF + 255)/256), blk, 0, stream>>>(w2, W2, Dd*DFF);

  // LN1 (+ zero pad rows) -> x bf16
  ln1_kernel<<<dim3((Bd*NPAD + 7)/8), blk, 0, stream>>>(hs, ln1w, ln1b, X_BF);

  // QKV projections (q scaled; v stored transposed per head), 32x64 macro-tiles
  {
    long tiles = (long)(Bd*NPAD/32) * (Dd/64);
    gemm_qkv<<<dim3((unsigned)((tiles + 7)/8), 3), blk, 0, stream>>>(
        X_BF, WQ, WK, WV, bq, bk, bv, Q_BF, K_BF, VT);
  }
  // scores -> d_out attn region (raw logits)
  {
    long tiles = (long)Bd * Hh * 13 * 13;
    gemm_scores<<<dim3((unsigned)((tiles + 7)/8)), blk, 0, stream>>>(Q_BF, K_BF, attn);
  }
  // softmax in place (fp32) + bf16 P with zero key-padding
  {
    long rows = (long)Bd * Hh * Nd;
    softmax_kernel<<<dim3((unsigned)((rows + 7)/8)), blk, 0, stream>>>(attn, P_BF);
  }
  // attn @ v -> attn_out bf16
  {
    long tiles = (long)Bd * Hh * 13 * 4;
    gemm_av<<<dim3((unsigned)((tiles + 7)/8)), blk, 0, stream>>>(P_BF, VT, AO);
  }
  // output projection + residual -> h_full fp32, 32x64 macro-tiles
  {
    long tiles = (long)(Bd*NPAD/32) * (Dd/64);
    gemm_o<<<dim3((unsigned)((tiles + 7)/8)), blk, 0, stream>>>(AO, WO, bo, hs, HFULL);
  }
  // CLS-attention ranking + prune-gather to 160 tokens
  prune_gather<<<dim3(Bd), blk, 0, stream>>>(attn, HFULL, HP);
  // cosine-sim merge: row norms, argmax, scatter-mean
  rownorm<<<dim3((Bd*PRUNE + 7)/8), blk, 0, stream>>>(HP, INV);
  sim_argmax<<<dim3(Bd*UNI), blk, 0, stream>>>(HP, INV, MS);
  merge_kernel<<<dim3(Bd), blk, 0, stream>>>(HP, MS, MERGED);
  // LN2 -> y bf16 (pad rows zeroed)
  ln2_kernel<<<dim3((Bd*MPAD2 + 7)/8), blk, 0, stream>>>(MERGED, ln2w, ln2b, YBF);
  // FFN1 (quick-GELU) -> ff bf16, 32x64 macro-tiles
  {
    long tiles = (long)(Bd*MPAD2/32) * (DFF/64);
    gemm_ffn1<<<dim3((unsigned)((tiles + 7)/8)), blk, 0, stream>>>(YBF, W1, b1, FF);
  }
  // FFN2 + residual -> d_out out region, 32x64 macro-tiles
  {
    long tiles = (long)(Bd*MPAD2/32) * (Dd/64);
    gemm_ffn2<<<dim3((unsigned)((tiles + 7)/8)), blk, 0, stream>>>(FF, W2, b2, MERGED, out_y);
  }
}
